// Attention_10342281248904
// MI455X (gfx1250) — compile-verified, matrix operand order checked by reading, not run on hardware
//
#include <hip/hip_runtime.h>
#include <hip/hip_bf16.h>
#include <cmath>
#include <cstdint>

// ---------------------------------------------------------------------------
// Problem constants (from the reference): R=128, N=256, B=1, E=512, H=8, M=4
// D = E/H = 64.
// ---------------------------------------------------------------------------
#define RR 128
#define NN 256
#define EE 512
#define HH 8
#define DD 64

typedef __bf16 bf16;
typedef bf16  v16bf __attribute__((ext_vector_type(16)));
typedef float v8f   __attribute__((ext_vector_type(8)));

union FragU { unsigned u[8]; v16bf v; };

// Wave-level LDS fence: all prior DS ops complete, compiler may not reorder
// memory ops across it.
__device__ __forceinline__ void lds_sync() {
    asm volatile("s_wait_dscnt 0" ::: "memory");
}
__device__ __forceinline__ void async_wait0() {
    asm volatile("s_wait_asynccnt 0x0" ::: "memory");
}
__device__ __forceinline__ void async_wait4() {
    asm volatile("s_wait_asynccnt 0x4" ::: "memory");
}

__device__ __forceinline__ unsigned pack_bf16(float a, float b) {
    union { bf16 h[2]; unsigned u; } t;
    t.h[0] = (bf16)a; t.h[1] = (bf16)b;
    return t.u;
}

// Stage a 16x32 f32 tile (row-major, leading dim ldg) into LDS as bf16.
// LDS tile layout: [16 rows][32 cols] bf16 == [16][16] u32 == 16 x 64B rows.
__device__ __forceinline__ void stage_f32(unsigned* lds, const float* g,
                                          int ldg, float scale, int lane) {
    int r = lane & 15, hf = lane >> 4;
    const float* s = g + r * ldg + hf * 16;
    unsigned* d = lds + r * 16 + hf * 8;
#pragma unroll
    for (int t = 0; t < 8; ++t)
        d[t] = pack_bf16(s[2 * t] * scale, s[2 * t + 1] * scale);
}

// Stage a 32x16 f32 tile stored [k][n] row-major into LDS transposed [n][k].
__device__ __forceinline__ void stage_f32_T(bf16* lds, const float* g,
                                            int ldg, int lane) {
    const float* s = g + lane * ldg;
#pragma unroll
    for (int n = 0; n < 16; ++n)
        lds[n * 32 + lane] = (bf16)s[n];
}

// Async-DMA a 16x32 bf16 tile (row-major, ldg elements, rows 16B-aligned)
// straight into LDS, bypassing VGPRs.  Per ISA 10.2 the generic pointer's low
// 32 bits are the LDS byte address; INST_OFFSET is added to BOTH the LDS and
// global addresses (ISA 15.18.3), so one VGPR pair + one LDS VGPR covers both
// 16B chunks per lane.  2 instructions/lane -> 2 ASYNCcnt per tile.
__device__ __forceinline__ void stage_async(unsigned lds_byte, const bf16* g,
                                            int ldg, int lane) {
    int r = lane & 15, hf = lane >> 4;
    unsigned la = lds_byte + r * 64 + hf * 16;
    unsigned long long ga =
        (unsigned long long)((const char*)(g + r * ldg) + hf * 16);
    asm volatile("global_load_async_to_lds_b128 %0, %1, off"
                 :: "v"(la), "v"(ga) : "memory");
    asm volatile("global_load_async_to_lds_b128 %0, %1, off offset:32"
                 :: "v"(la), "v"(ga) : "memory");
}

// Load a WMMA bf16 fragment from an LDS [16][32] bf16 tile.
// ISA 16-bit A/B layout: lane L -> row = L%16, half = L/16;
// VGPR v holds K pair at kbase = (v<4 ? 2v : 2v+8) + 8*half.
__device__ __forceinline__ v16bf load_frag(const unsigned* lds, int lane) {
    int row = lane & 15, hf = lane >> 4;
    FragU f;
#pragma unroll
    for (int v = 0; v < 8; ++v) {
        int kb = (v < 4 ? 2 * v : 2 * v + 8) + 8 * hf;  // even
        f.u[v] = lds[row * 16 + (kb >> 1)];
    }
    return f.v;
}

__device__ __forceinline__ v8f wmma_bf16(v16bf a, v16bf b, v8f c) {
    return __builtin_amdgcn_wmma_f32_16x16x32_bf16(false, a, false, b,
                                                   (short)0, c, false, false);
}

// Packed bf16 scale: <16 x bf16> * splat -> v_pk_mul_bf16, co-executes with
// the XDL WMMA pipe.
__device__ __forceinline__ v16bf scale_frag(v16bf b, float s) {
    bf16 h = (bf16)s;
    v16bf sv;
#pragma unroll
    for (int t = 0; t < 16; ++t) sv[t] = h;
    return b * sv;
}

// ---------------------------------------------------------------------------
// k1: fused Q/K/V projection.  X[32768,512] @ {Wq,Wk,Wv}[512,512] (+bias).
// Scatters:
//   q_t[h][d][i][r] = (xWq+bq)*scaling   (bf16)   -- GEMM-A layout for k2
//   k_t[h][d][j][r] = (xWk+bk)           (bf16)   -- GEMM-B layout for k2
//   v_t[h][r*64+d][j] = (xWv+bv)         (bf16)   -- GEMM-B layout for k4
// One wave computes one 16x16 tile for all three outputs, sharing the A stage.
// ---------------------------------------------------------------------------
__global__ void __launch_bounds__(256)
k1_proj(const float* __restrict__ x,
        const float* __restrict__ Wq, const float* __restrict__ bq,
        const float* __restrict__ Wk, const float* __restrict__ bk,
        const float* __restrict__ Wv, const float* __restrict__ bv,
        bf16* __restrict__ q_t, bf16* __restrict__ k_t, bf16* __restrict__ v_t,
        float scaling) {
    __shared__ unsigned lds[8][4 * 256];  // per wave: A, Bq, Bk, Bv
    int lane = threadIdx.x & 31, wav = threadIdx.x >> 5;
    int gw = blockIdx.x * 8 + wav;        // 65536 wave-tiles
    int m0 = (gw >> 5) * 16;              // row tile (r*N+i)
    int n0 = (gw & 31) * 16;              // col tile (e)
    unsigned* A  = lds[wav];
    unsigned* Bq = A + 256;
    unsigned* Bk = A + 512;
    unsigned* Bv = A + 768;
    v8f cq = {}, ck = {}, cv = {};
    for (int k0 = 0; k0 < EE; k0 += 32) {
        if (k0 + 32 < EE) {  // speculative prefetch of next k-step tiles
            __builtin_prefetch(x + m0 * EE + k0 + 32, 0, 0);
            __builtin_prefetch(Wq + (size_t)(k0 + 32) * EE + n0, 0, 0);
            __builtin_prefetch(Wk + (size_t)(k0 + 32) * EE + n0, 0, 0);
            __builtin_prefetch(Wv + (size_t)(k0 + 32) * EE + n0, 0, 0);
        }
        stage_f32(A, x + m0 * EE + k0, EE, 1.f, lane);
        stage_f32_T((bf16*)Bq, Wq + (size_t)k0 * EE + n0, EE, lane);
        stage_f32_T((bf16*)Bk, Wk + (size_t)k0 * EE + n0, EE, lane);
        stage_f32_T((bf16*)Bv, Wv + (size_t)k0 * EE + n0, EE, lane);
        lds_sync();
        v16bf a  = load_frag(A, lane);
        v16bf fq = load_frag(Bq, lane);
        v16bf fk = load_frag(Bk, lane);
        v16bf fv = load_frag(Bv, lane);
        cq = wmma_bf16(a, fq, cq);
        ck = wmma_bf16(a, fk, ck);
        cv = wmma_bf16(a, fv, cv);
        lds_sync();  // loads done before next iteration's stores (WAR)
    }
    int hf = lane >> 4, nl = lane & 15;
    int e = n0 + nl, h = e >> 6, d = e & 63;
#pragma unroll
    for (int i = 0; i < 8; ++i) {
        int m = m0 + i + 8 * hf;
        int r = m >> 8, ii = m & 255;
        q_t[(size_t)e * (NN * RR) + ii * RR + r] = (bf16)((cq[i] + bq[e]) * scaling);
        k_t[(size_t)e * (NN * RR) + ii * RR + r] = (bf16)(ck[i] + bk[e]);
        v_t[((size_t)h * (RR * DD) + r * DD + d) * NN + ii] = (bf16)(cv[i] + bv[e]);
    }
}

// ---------------------------------------------------------------------------
// k2: qk contraction, fused with attn_weights=sum_d and q1=(qk@Wq1).mean(h).
// Grid: 256 blocks = (ti,tj) 16x16-tile pairs; 8 waves/block, wave = head h.
// Double-buffered async LDS staging: issue step s+1's tiles (4 ASYNCcnt)
// into the other buffer, then wait asynccnt<=4 so step s's group (in-order
// completion) is resident.  9 WMMAs per 32-wide k-step.
// ---------------------------------------------------------------------------
__global__ void __launch_bounds__(256)
k2_qk(const bf16* __restrict__ q_t, const bf16* __restrict__ k_t,
      const float* __restrict__ Wq1, const float* __restrict__ bq1,
      float* __restrict__ attnw, float* __restrict__ q1) {
    __shared__ unsigned lds[8][2][512];  // [wave][buf][A(256)|B(256)]
    __shared__ float q1red[16 * 16 * 8];
    int lane = threadIdx.x & 31, wav = threadIdx.x >> 5;
    int h = wav;
    int i0 = (blockIdx.x >> 4) * 16, j0 = (blockIdx.x & 15) * 16;
    for (int t = threadIdx.x; t < 2048; t += 256) q1red[t] = 0.f;
    __syncthreads();
    unsigned base = (unsigned)(uintptr_t)(&lds[wav][0][0]);
    const bf16* qb0 = q_t + (size_t)h * DD * (NN * RR) + i0 * RR;
    const bf16* kb0 = k_t + (size_t)h * DD * (NN * RR) + j0 * RR;

    auto issue = [&](int s, int buf) {
        int d = s >> 2, ks = (s & 3) << 5;
        stage_async(base + buf * 2048,
                    qb0 + (size_t)d * (NN * RR) + ks, RR, lane);
        stage_async(base + buf * 2048 + 1024,
                    kb0 + (size_t)d * (NN * RR) + ks, RR, lane);
    };

    v8f accA = {};
    v8f accQ[8];
#pragma unroll
    for (int c = 0; c < 8; ++c) { v8f z = {}; accQ[c] = z; }

    issue(0, 0);
    for (int s = 0; s < DD * 4; ++s) {          // 256 k-steps of 32
        int buf = s & 1;
        if (s + 1 < DD * 4) { issue(s + 1, buf ^ 1); async_wait4(); }
        else                { async_wait0(); }
        const unsigned* A = &lds[wav][buf][0];
        const unsigned* B = A + 256;
        v16bf a = load_frag(A, lane);
        v16bf b = load_frag(B, lane);
        accA = wmma_bf16(a, b, accA);
        int d = s >> 2;
#pragma unroll
        for (int c = 0; c < 8; ++c)
            accQ[c] = wmma_bf16(a, scale_frag(b, Wq1[d * 8 + c]), accQ[c]);
        lds_sync();  // frag reads retired before this buf is re-DMA'd
    }
    int hf = lane >> 4, nl = lane & 15;
#pragma unroll
    for (int i = 0; i < 8; ++i) {
        int gi = i0 + i + 8 * hf, gj = j0 + nl;
        attnw[(size_t)h * (NN * NN) + gi * NN + gj] = accA[i];
    }
#pragma unroll
    for (int c = 0; c < 8; ++c)
#pragma unroll
        for (int i = 0; i < 8; ++i)
            atomicAdd(&q1red[((i + 8 * hf) * 16 + nl) * 8 + c], accQ[c][i]);
    __syncthreads();
    for (int t = threadIdx.x; t < 2048; t += 256) {
        int c = t & 7, n_ = (t >> 3) & 15, m_ = t >> 7;
        q1[((size_t)(i0 + m_) * NN + (j0 + n_)) * 8 + c] =
            q1red[t] * (1.f / HH) + bq1[c];
    }
}

// ---------------------------------------------------------------------------
// k3: network-MLP bias, eye mask, row softmax.  One block per (h, i) row.
// Writes attn_probs directly into its slot of d_out; block 0 copies l.
// ---------------------------------------------------------------------------
__global__ void __launch_bounds__(256)
k3_softmax(const float* __restrict__ attnw, const float* __restrict__ q1,
           const float* __restrict__ network, const float* __restrict__ Wk1,
           const float* __restrict__ bk1, const float* __restrict__ l,
           float* __restrict__ probs_out, float* __restrict__ l_out) {
    int h = blockIdx.x >> 8, i = blockIdx.x & 255;
    int j = threadIdx.x;
    float q1v[8];
#pragma unroll
    for (int c = 0; c < 8; ++c) q1v[c] = q1[((size_t)i * NN + j) * 8 + c];
    float na[4], nwv[4];
#pragma unroll
    for (int m = 0; m < 4; ++m) {
        float nw = network[((size_t)i * NN + j) * 4 + m];
        nwv[m] = nw;
        float s = 0.f;
#pragma unroll
        for (int c = 0; c < 8; ++c) s += q1v[c] * (nw * Wk1[c] + bk1[c]);
        na[m] = s;
    }
    float mx = fmaxf(fmaxf(na[0], na[1]), fmaxf(na[2], na[3]));
    float es = 0.f, ex[4];
#pragma unroll
    for (int m = 0; m < 4; ++m) { ex[m] = __expf(na[m] - mx); es += ex[m]; }
    float nb = 0.f;
#pragma unroll
    for (int m = 0; m < 4; ++m) nb += nwv[m] * (ex[m] / es);
    float w = attnw[(size_t)h * (NN * NN) + (size_t)i * NN + j] + l[h] * nb;
    if (j == i) w = -1e9f;
    __shared__ float red[256];
    red[j] = w; __syncthreads();
    for (int s = 128; s > 0; s >>= 1) {
        if (j < s) red[j] = fmaxf(red[j], red[j + s]);
        __syncthreads();
    }
    float rmax = red[0]; __syncthreads();
    float e = __expf(w - rmax);
    red[j] = e; __syncthreads();
    for (int s = 128; s > 0; s >>= 1) {
        if (j < s) red[j] += red[j + s];
        __syncthreads();
    }
    probs_out[(size_t)h * (NN * NN) + (size_t)i * NN + j] = e / red[0];
    if (blockIdx.x == 0 && j < HH) l_out[j] = l[j];
}

// ---------------------------------------------------------------------------
// k4: context = P[h] @ V[h].   Per h: [256 x 8192 x 256] GEMM.
// B operand (bf16 V) is async-DMA'd into LDS; A (f32 probs) staged via VGPRs.
// Output scattered as bf16 ctx[(r*N+i)*E + h*64+d] for the final projection.
// ---------------------------------------------------------------------------
__global__ void __launch_bounds__(256)
k4_ctx(const float* __restrict__ probs, const bf16* __restrict__ v_t,
       bf16* __restrict__ ctx) {
    __shared__ unsigned lds[8][512];
    int lane = threadIdx.x & 31, wav = threadIdx.x >> 5;
    int gw = blockIdx.x * 8 + wav;            // 8 * 16 * 512 tiles
    int h = gw >> 13;
    int rem = gw & 8191;
    int i0 = (rem >> 9) * 16;
    int n0 = (rem & 511) * 16;                // rd tile
    unsigned* A = lds[wav];
    unsigned baseB = (unsigned)(uintptr_t)(A + 256);
    v8f acc = {};
    const float* Ab = probs + (size_t)h * (NN * NN) + i0 * NN;
    const bf16* Bb = v_t + (size_t)h * (RR * DD) * NN + (size_t)n0 * NN;
    for (int k0 = 0; k0 < NN; k0 += 32) {
        stage_async(baseB, Bb + k0, NN, lane);
        stage_f32(A, Ab + k0, NN, 1.f, lane);
        async_wait0();
        lds_sync();
        acc = wmma_bf16(load_frag(A, lane), load_frag(A + 256, lane), acc);
        lds_sync();
    }
    int hf = lane >> 4, nl = lane & 15;
    int rd = n0 + nl, r = rd >> 6, d = rd & 63;
#pragma unroll
    for (int i = 0; i < 8; ++i) {
        int gi = i0 + i + 8 * hf;
        ctx[((size_t)r * NN + gi) * EE + h * DD + d] = (bf16)acc[i];
    }
}

// ---------------------------------------------------------------------------
// k5: output projection: ctx[32768,512] @ Wo[512,512] + bo -> d_out.
// A operand (bf16 ctx) async-DMA'd; B (f32 Wo) transpose-staged via VGPRs.
// ---------------------------------------------------------------------------
__global__ void __launch_bounds__(256)
k5_out(const bf16* __restrict__ ctx, const float* __restrict__ Wo,
       const float* __restrict__ bo, float* __restrict__ out) {
    __shared__ unsigned lds[8][512];
    int lane = threadIdx.x & 31, wav = threadIdx.x >> 5;
    int gw = blockIdx.x * 8 + wav;
    int m0 = (gw >> 5) * 16, n0 = (gw & 31) * 16;
    unsigned* A = lds[wav];
    unsigned baseA = (unsigned)(uintptr_t)A;
    unsigned* B = A + 256;
    v8f acc = {};
    for (int k0 = 0; k0 < EE; k0 += 32) {
        if (k0 + 32 < EE)
            __builtin_prefetch(Wo + (size_t)(k0 + 32) * EE + n0, 0, 0);
        stage_async(baseA, ctx + (size_t)m0 * EE + k0, EE, lane);
        stage_f32_T((bf16*)B, Wo + (size_t)k0 * EE + n0, EE, lane);
        async_wait0();
        lds_sync();
        acc = wmma_bf16(load_frag(A, lane), load_frag(B, lane), acc);
        lds_sync();
    }
    int hf = lane >> 4, nl = lane & 15;
    int e = n0 + nl;
    float bias = bo[e];
#pragma unroll
    for (int i = 0; i < 8; ++i) {
        int m = m0 + i + 8 * hf;
        out[(size_t)m * EE + e] = acc[i] + bias;
    }
}

// ---------------------------------------------------------------------------
extern "C" void kernel_launch(void* const* d_in, const int* in_sizes, int n_in,
                              void* d_out, int out_size, void* d_ws,
                              size_t ws_size, hipStream_t stream) {
    const float* x       = (const float*)d_in[0];
    const float* network = (const float*)d_in[1];
    const float* Wq  = (const float*)d_in[2];  const float* bq  = (const float*)d_in[3];
    const float* Wk  = (const float*)d_in[4];  const float* bk  = (const float*)d_in[5];
    const float* Wv  = (const float*)d_in[6];  const float* bv  = (const float*)d_in[7];
    const float* Wo  = (const float*)d_in[8];  const float* bo  = (const float*)d_in[9];
    const float* Wq1 = (const float*)d_in[10]; const float* bq1 = (const float*)d_in[11];
    const float* Wk1 = (const float*)d_in[12]; const float* bk1 = (const float*)d_in[13];
    const float* l   = (const float*)d_in[14];

    const size_t nBig = (size_t)RR * NN * EE;  // 16,777,216 elements
    char* ws = (char*)d_ws;
    bf16* q_t = (bf16*)ws;             ws += nBig * sizeof(bf16);
    bf16* k_t = (bf16*)ws;             ws += nBig * sizeof(bf16);
    bf16* v_t = (bf16*)ws;             ws += nBig * sizeof(bf16);
    bf16* ctx = (bf16*)ws;             ws += nBig * sizeof(bf16);
    float* attnw = (float*)ws;         ws += (size_t)HH * NN * NN * sizeof(float);
    float* q1    = (float*)ws;         ws += (size_t)NN * NN * 8 * sizeof(float);

    float* out_main  = (float*)d_out;
    float* probs_out = out_main + nBig;
    float* l_out     = probs_out + (size_t)HH * NN * NN;

    const float scaling = (1.0f / 8.0f) / sqrtf((float)RR);  // D^-0.5 / sqrt(R)

    k1_proj<<<8192, 256, 0, stream>>>(x, Wq, bq, Wk, bk, Wv, bv,
                                      q_t, k_t, v_t, scaling);
    k2_qk<<<256, 256, 0, stream>>>(q_t, k_t, Wq1, bq1, attnw, q1);
    k3_softmax<<<HH * NN, 256, 0, stream>>>(attnw, q1, network, Wk1, bk1, l,
                                            probs_out, l_out);
    k4_ctx<<<8192, 256, 0, stream>>>(probs_out, v_t, ctx);
    k5_out<<<8192, 256, 0, stream>>>(ctx, Wo, bo, out_main);
}
